// GraphClassificationModel_78254304133858
// MI455X (gfx1250) — matrix operand order repaired and checked by reading, not compiled, not run on hardware
//
#include <hip/hip_runtime.h>

typedef __attribute__((ext_vector_type(16))) _Float16 v16h;
typedef __attribute__((ext_vector_type(8)))  float    v8f;

#define TB 256
#define NEG_SLOPE 0.2f

// ---------------------------------------------------------------------------
// WMMA GEMM: Y[nrows x 64] = H[nrows x cin] @ W[cin x 64] + b
// Block = 256 threads = 8 waves; each wave computes a 16x64 row-slab.
// Tiles are staged in LDS in *fragment order*: each lane's 16 f16 operand
// values are contiguous + 32B aligned, so a fragment load = 2x ds_load_b128.
// Fragment packing (16-bit A 16x32 / B 32x16), contiguous k order shared by A and B:
//   A: lane<16 -> M=lane,  K=0..15; lane>=16 -> K=16..31
//   B: lane<16 -> K=0..15; lane>=16 -> K=16..31; N = lane%16
// ---------------------------------------------------------------------------
__global__ __launch_bounds__(256) void gemm_n64_kernel(
    const float* __restrict__ H, int nrows, int cin,
    const float* __restrict__ W, const float* __restrict__ bvec,
    float* __restrict__ Y)
{
    __shared__ v16h lwB[2 * 4 * 32];   // [ks][tile][lane]  : 8 KB
    __shared__ v16h lhA[8 * 2 * 32];   // [wave][ks][lane]  : 16 KB
    _Float16* lwh = (_Float16*)lwB;
    _Float16* lhh = (_Float16*)lhA;

    const int tid    = threadIdx.x;
    const int rowblk = blockIdx.x * 128;

    // Stage W (coalesced global read, fragment-order LDS write, zero-pad K)
    for (int idx = tid; idx < 64 * 64; idx += TB) {
        int k = idx >> 6, n = idx & 63;
        float v  = (k < cin) ? W[idx] : 0.0f;          // W[k*64+n] == W[idx]
        int ks   = k >> 5, kin = k & 31;
        int lane = ((kin >> 4) << 4) | (n & 15);
        int j    = kin & 15;
        int t    = n >> 4;
        lwh[((((ks << 2) + t) << 5) + lane) * 16 + j] = (_Float16)v;
    }
    // Stage H tile (coalesced global read when cin==64, fragment-order write)
    for (int idx = tid; idx < 128 * 64; idx += TB) {
        int r = idx >> 6, k = idx & 63;
        int row = rowblk + r;
        float v  = (row < nrows && k < cin) ? H[(size_t)row * cin + k] : 0.0f;
        int wv   = r >> 4, l16 = r & 15;
        int ks   = k >> 5, kin = k & 31;
        int hb   = kin >> 4;
        int j    = kin & 15;
        int lane = (hb << 4) | l16;
        lhh[((((wv << 1) + ks) << 5) + lane) * 16 + j] = (_Float16)v;
    }
    __syncthreads();

    const int wave = tid >> 5;
    const int lane = tid & 31;
    const int hb   = lane >> 4;
    const int l16  = lane & 15;
    const int ksteps = (cin + 31) >> 5;

    v8f acc[4];
#pragma unroll
    for (int t = 0; t < 4; ++t) acc[t] = (v8f){0.f,0.f,0.f,0.f,0.f,0.f,0.f,0.f};

    for (int ks = 0; ks < ksteps; ++ks) {
        v16h a = lhA[(((wave << 1) + ks) << 5) + lane];   // 2x ds_load_b128
#pragma unroll
        for (int t = 0; t < 4; ++t) {
            v16h b = lwB[(((ks << 2) + t) << 5) + lane];  // 2x ds_load_b128
            acc[t] = __builtin_amdgcn_wmma_f32_16x16x32_f16(
                false, a, false, b, (short)0, acc[t], false, false);
        }
    }

    // D layout: VGPR r -> row M = r + 8*hb, col N = lane%16
#pragma unroll
    for (int t = 0; t < 4; ++t) {
        int col = t * 16 + l16;
        float bv = bvec[col];
#pragma unroll
        for (int r = 0; r < 8; ++r) {
            int row = rowblk + wave * 16 + r + hb * 8;
            if (row < nrows) Y[(size_t)row * 64 + col] = acc[t][r] + bv;
        }
    }
}

// ---------------------------------------------------------------------------
// Per-layer init: segment-max keys, denominators, aggregate buffer
// ---------------------------------------------------------------------------
__global__ __launch_bounds__(256) void init_layer_kernel(
    unsigned* __restrict__ mkey, float* __restrict__ denom,
    float* __restrict__ agg, int N)
{
    int i = blockIdx.x * TB + threadIdx.x;
    if (i < N * 64) agg[i] = 0.0f;
    if (i < N) { mkey[i] = 0u; denom[i] = 0.0f; }
}

__device__ __forceinline__ unsigned f2key(float f) {
    unsigned u = __float_as_uint(f);
    return (u & 0x80000000u) ? ~u : (u | 0x80000000u);
}
__device__ __forceinline__ float key2f(unsigned k) {
    return __uint_as_float((k & 0x80000000u) ? (k ^ 0x80000000u) : ~k);
}

// ---------------------------------------------------------------------------
// Edge scores + atomic segment max.  8 lanes per edge, float4 (128b) loads.
// ---------------------------------------------------------------------------
__global__ __launch_bounds__(256) void edge_score_kernel(
    const float* __restrict__ XL, const float* __restrict__ XR,
    const long long* __restrict__ ei, int E, int N,
    const float* __restrict__ att, float* __restrict__ esc,
    unsigned* __restrict__ mkey)
{
    int gtid = blockIdx.x * TB + threadIdx.x;
    int eid  = gtid >> 3;
    int sub  = gtid & 7;
    int Etot = E + N;
    if (eid >= Etot) return;

    int src, dst;
    if (eid < E) { src = (int)ei[eid]; dst = (int)ei[(size_t)E + eid]; }
    else         { src = dst = eid - E; }

    const float4* pl = (const float4*)(XL + (size_t)src * 64 + sub * 8);
    const float4* pr = (const float4*)(XR + (size_t)dst * 64 + sub * 8);
    float4 l0 = pl[0], l1 = pl[1];
    float4 r0 = pr[0], r1 = pr[1];

    float vals[8] = { l0.x + r0.x, l0.y + r0.y, l0.z + r0.z, l0.w + r0.w,
                      l1.x + r1.x, l1.y + r1.y, l1.z + r1.z, l1.w + r1.w };
    const float* ap = att + sub * 8;
    float s = 0.0f;
#pragma unroll
    for (int c = 0; c < 8; ++c) {
        float v = vals[c];
        v = (v > 0.0f) ? v : NEG_SLOPE * v;
        s += ap[c] * v;
    }
    // reduce across the 8-lane group (groups are xor-aligned inside wave32)
    s += __shfl_xor(s, 1, 32);
    s += __shfl_xor(s, 2, 32);
    s += __shfl_xor(s, 4, 32);

    if (sub == 0) {
        esc[eid] = s;
        atomicMax(&mkey[dst], f2key(s));
    }
}

// ---------------------------------------------------------------------------
// Edge aggregation: p = exp(e - m[dst]); atomic denom + 64-ch scatter-add
// ---------------------------------------------------------------------------
__global__ __launch_bounds__(256) void edge_agg_kernel(
    const float* __restrict__ XL, const float* __restrict__ esc,
    const long long* __restrict__ ei, int E, int N,
    const unsigned* __restrict__ mkey, float* __restrict__ denom,
    float* __restrict__ agg)
{
    int gtid = blockIdx.x * TB + threadIdx.x;
    int eid  = gtid >> 3;
    int sub  = gtid & 7;
    int Etot = E + N;
    if (eid >= Etot) return;

    int src, dst;
    if (eid < E) { src = (int)ei[eid]; dst = (int)ei[(size_t)E + eid]; }
    else         { src = dst = eid - E; }

    float m = key2f(mkey[dst]);
    float p = __expf(esc[eid] - m);
    if (sub == 0) atomicAdd(&denom[dst], p);

    const float4* pl = (const float4*)(XL + (size_t)src * 64 + sub * 8);
    float4 l0 = pl[0], l1 = pl[1];
    float* ad = agg + (size_t)dst * 64 + sub * 8;
    atomicAdd(ad + 0, p * l0.x); atomicAdd(ad + 1, p * l0.y);
    atomicAdd(ad + 2, p * l0.z); atomicAdd(ad + 3, p * l0.w);
    atomicAdd(ad + 4, p * l1.x); atomicAdd(ad + 5, p * l1.y);
    atomicAdd(ad + 6, p * l1.z); atomicAdd(ad + 7, p * l1.w);
}

// ---------------------------------------------------------------------------
// Finalize: h = agg/denom + bias (+ relu between layers)
// ---------------------------------------------------------------------------
__global__ __launch_bounds__(256) void finalize_kernel(
    const float* __restrict__ agg, const float* __restrict__ denom,
    const float* __restrict__ bias, float* __restrict__ hout,
    int N, int do_relu)
{
    int i = blockIdx.x * TB + threadIdx.x;
    if (i >= N * 64) return;
    int n = i >> 6, c = i & 63;
    float v = agg[i] / denom[n] + bias[c];
    if (do_relu) v = fmaxf(v, 0.0f);
    hout[i] = v;
}

__global__ __launch_bounds__(256) void zero_kernel(float* __restrict__ p, int n) {
    int i = blockIdx.x * TB + threadIdx.x;
    if (i < n) p[i] = 0.0f;
}

// global_add_pool via atomics
__global__ __launch_bounds__(256) void pool_kernel(
    const float* __restrict__ h, const long long* __restrict__ batch,
    float* __restrict__ pooled, int N)
{
    int i = blockIdx.x * TB + threadIdx.x;
    if (i >= N * 64) return;
    int n = i >> 6, c = i & 63;
    atomicAdd(&pooled[(size_t)batch[n] * 64 + c], h[i]);
}

// ---------------------------------------------------------------------------
// Final FC on WMMA: out[G x OUT] = pooled[G x 64] @ fc_w[64 x OUT] + fc_b
// One block (8 waves) per 16-column strip; wave w does row-tile w.
// W strip + pooled staged in fragment-order LDS (wide ds_load_b128 feeds).
// ---------------------------------------------------------------------------
__global__ __launch_bounds__(256) void fc_kernel(
    const float* __restrict__ P, const float* __restrict__ W,
    const float* __restrict__ b, float* __restrict__ out, int G, int OUT)
{
    __shared__ v16h lw[2 * 32];        // [ks][lane]        : 2 KB
    __shared__ v16h lp[8 * 2 * 32];    // [wave][ks][lane]  : 16 KB
    _Float16* lwh = (_Float16*)lw;
    _Float16* lph = (_Float16*)lp;

    const int tid  = threadIdx.x;
    const int tcol = blockIdx.x;       // 16-column strip index

    // Stage W[0..63][tcol*16 .. tcol*16+15] in B-fragment order
    for (int idx = tid; idx < 64 * 16; idx += TB) {
        int k = idx >> 4, n = idx & 15;
        float v  = W[(size_t)k * OUT + tcol * 16 + n];
        int ks   = k >> 5, kin = k & 31;
        int lane = ((kin >> 4) << 4) | n;
        int j    = kin & 15;
        lwh[((ks << 5) + lane) * 16 + j] = (_Float16)v;
    }
    // Stage pooled [G x 64] in A-fragment order (8 row-tiles)
    for (int idx = tid; idx < 128 * 64; idx += TB) {
        int r = idx >> 6, k = idx & 63;
        float v  = (r < G) ? P[(size_t)r * 64 + k] : 0.0f;
        int wv   = r >> 4, l16 = r & 15;
        int ks   = k >> 5, kin = k & 31;
        int hb   = kin >> 4;
        int j    = kin & 15;
        int lane = (hb << 4) | l16;
        lph[((((wv << 1) + ks) << 5) + lane) * 16 + j] = (_Float16)v;
    }
    __syncthreads();

    const int wave = tid >> 5;
    const int lane = tid & 31;
    const int hb   = lane >> 4;
    const int l16  = lane & 15;

    v8f acc = (v8f){0.f,0.f,0.f,0.f,0.f,0.f,0.f,0.f};
#pragma unroll
    for (int ks = 0; ks < 2; ++ks) {
        v16h a = lp[(((wave << 1) + ks) << 5) + lane];
        v16h bf = lw[(ks << 5) + lane];
        acc = __builtin_amdgcn_wmma_f32_16x16x32_f16(
            false, a, false, bf, (short)0, acc, false, false);
    }
    int col = tcol * 16 + l16;
    float bv = b[col];
#pragma unroll
    for (int r = 0; r < 8; ++r) {
        int row = wave * 16 + r + hb * 8;
        if (row < G) out[(size_t)row * OUT + col] = acc[r] + bv;
    }
}

// ---------------------------------------------------------------------------
extern "C" void kernel_launch(void* const* d_in, const int* in_sizes, int n_in,
                              void* d_out, int out_size, void* d_ws, size_t ws_size,
                              hipStream_t stream) {
    const int NLYR = 6, C = 64, OUT = 2048;
    const float*      x     = (const float*)d_in[0];
    const long long*  ei    = (const long long*)d_in[1];
    const long long*  batch = (const long long*)d_in[2];

    const int E    = in_sizes[1] / 2;
    const int N    = in_sizes[2];
    const int cin0 = in_sizes[0] / N;       // 3
    const int G    = out_size / OUT;        // 128

    const float *Wl[6], *Wr[6], *bl[6], *br[6], *att[6], *bias[6], *fcw, *fcb;
    if (n_in >= 41) {   // each python list element is its own input tensor
        for (int i = 0; i < 6; ++i) {
            Wl[i]   = (const float*)d_in[3 + i];
            Wr[i]   = (const float*)d_in[9 + i];
            bl[i]   = (const float*)d_in[15 + i];
            br[i]   = (const float*)d_in[21 + i];
            att[i]  = (const float*)d_in[27 + i];
            bias[i] = (const float*)d_in[33 + i];
        }
        fcw = (const float*)d_in[39];
        fcb = (const float*)d_in[40];
    } else {            // lists concatenated flat per dict key
        const float* pWl = (const float*)d_in[3];
        const float* pWr = (const float*)d_in[4];
        const float* pbl = (const float*)d_in[5];
        const float* pbr = (const float*)d_in[6];
        const float* pat = (const float*)d_in[7];
        const float* pbi = (const float*)d_in[8];
        size_t wo = 0;
        for (int i = 0; i < 6; ++i) {
            Wl[i] = pWl + wo; Wr[i] = pWr + wo;
            wo += (size_t)(i == 0 ? cin0 : C) * C;
            bl[i]   = pbl + (size_t)i * C;
            br[i]   = pbr + (size_t)i * C;
            att[i]  = pat + (size_t)i * C;
            bias[i] = pbi + (size_t)i * C;
        }
        fcw = (const float*)d_in[9];
        fcb = (const float*)d_in[10];
    }

    // Workspace carve-out
    char* w = (char*)d_ws;
    const size_t N64 = (size_t)N * 64 * sizeof(float);
    auto algn = [](size_t v) { return (v + 255) & ~(size_t)255; };
    float*    xl     = (float*)w;    w += algn(N64);
    float*    xr     = (float*)w;    w += algn(N64);
    float*    agg    = (float*)w;    w += algn(N64);
    float*    hA     = (float*)w;    w += algn(N64);
    float*    hB     = (float*)w;    w += algn(N64);
    float*    denom  = (float*)w;    w += algn((size_t)N * 4);
    unsigned* mkey   = (unsigned*)w; w += algn((size_t)N * 4);
    float*    esc    = (float*)w;    w += algn((size_t)(E + N) * 4);
    float*    pooled = (float*)w;    w += algn((size_t)G * 64 * 4);
    (void)ws_size;

    const int Etot  = E + N;
    const dim3 blk(TB);
    const int gGemm = (N + 127) / 128;
    const int gNC   = (N * 64 + TB - 1) / TB;
    const int gEdge = (Etot * 8 + TB - 1) / TB;

    const float* h = x;
    int cin = cin0;
    float* hbuf[2] = { hA, hB };
    for (int l = 0; l < NLYR; ++l) {
        gemm_n64_kernel<<<gGemm, blk, 0, stream>>>(h, N, cin, Wl[l], bl[l], xl);
        gemm_n64_kernel<<<gGemm, blk, 0, stream>>>(h, N, cin, Wr[l], br[l], xr);
        init_layer_kernel<<<gNC, blk, 0, stream>>>(mkey, denom, agg, N);
        edge_score_kernel<<<gEdge, blk, 0, stream>>>(xl, xr, ei, E, N, att[l], esc, mkey);
        edge_agg_kernel<<<gEdge, blk, 0, stream>>>(xl, esc, ei, E, N, mkey, denom, agg);
        float* ho = hbuf[l & 1];
        finalize_kernel<<<gNC, blk, 0, stream>>>(agg, denom, bias[l], ho, N, l < NLYR - 1);
        h = ho;
        cin = C;
    }

    zero_kernel<<<(G * 64 + TB - 1) / TB, blk, 0, stream>>>(pooled, G * 64);
    pool_kernel<<<gNC, blk, 0, stream>>>(h, batch, pooled, N);

    fc_kernel<<<OUT / 16, blk, 0, stream>>>(pooled, fcw, fcb, (float*)d_out, G, OUT);
}
